// GCNGenerator_61151744361123
// MI455X (gfx1250) — matrix operand order repaired and checked by reading, not compiled, not run on hardware
//
#include <hip/hip_runtime.h>
#include <hip/hip_bf16.h>

typedef __attribute__((ext_vector_type(16))) _Float16 v16h;
typedef __attribute__((ext_vector_type(8)))  _Float16 v8h;
typedef __attribute__((ext_vector_type(8)))  float    v8f;

#define FIN 128  // K dimension for all three GEMMs

// ---------------------------------------------------------------------------
// Weight prep: Wt[n*K + k] = (f16) W[k*N + n]   (transpose + f32->f16)
// ---------------------------------------------------------------------------
__global__ void wconv_kernel(const float* __restrict__ W, _Float16* __restrict__ Wt,
                             int K, int N) {
    int idx = blockIdx.x * blockDim.x + threadIdx.x;
    if (idx >= K * N) return;
    int n = idx / K, k = idx % K;
    Wt[idx] = (_Float16)W[k * N + n];
}

// ---------------------------------------------------------------------------
// Degree / normalization
// ---------------------------------------------------------------------------
__global__ void deg_init_kernel(float* __restrict__ deg, int n) {
    int i = blockIdx.x * blockDim.x + threadIdx.x;
    if (i < n) deg[i] = 1.0f;  // self loop
}
__global__ void deg_count_kernel(const int* __restrict__ dst, float* __restrict__ deg, int E) {
    int e = blockIdx.x * blockDim.x + threadIdx.x;
    if (e < E) atomicAdd(&deg[dst[e]], 1.0f);
}
__global__ void deg_rsqrt_kernel(float* __restrict__ d, int n) {
    int i = blockIdx.x * blockDim.x + threadIdx.x;
    if (i < n) d[i] = rsqrtf(d[i]);   // in place: deg -> deg^{-1/2}
}

// ---------------------------------------------------------------------------
// WMMA GEMM: C[M x NT] = A[M x 128] (f32, cvt f16) @ W[128 x NT]  (+bias)
// Wt = f16 transposed weight [NT x 128] row-major, L2-resident.
// Block = 16 rows x NT cols (NT/16 waves). REQUIRES M % 16 == 0 (M = 100000).
// All indexing 32-bit; epilogue staged through LDS for linear b128 stores.
// ---------------------------------------------------------------------------
template <int NT, bool BIAS>
__global__ __launch_bounds__(NT * 2)
void gemm_wmma_kernel(const float* __restrict__ A,
                      const _Float16* __restrict__ Wt,
                      const float* __restrict__ bias,
                      float* __restrict__ C) {
    __shared__ __align__(32) _Float16 As[16 * FIN];   // 4 KB A tile (f16)
    __shared__ __align__(32) float    Cs[16 * NT];    // 8/4 KB C tile (f32)

    const int tid  = threadIdx.x;
    const int wave = tid >> 5;
    const int lane = tid & 31;
    const int row_base = blockIdx.x * 16;

    // --- Cooperative A-tile load: contiguous 2048 floats -> f16 LDS ---
    const int nthreads = NT * 2;
    const int per      = 2048 / nthreads;   // 8 (NT=128) or 16 (NT=64)
    const float* Ab = A + row_base * FIN;
#pragma unroll
    for (int c8 = 0; c8 < per / 8; ++c8) {
        int e = tid * per + c8 * 8;
        float4 u0 = *(const float4*)&Ab[e];
        float4 u1 = *(const float4*)&Ab[e + 4];
        v8h h;
        h[0] = (_Float16)u0.x; h[1] = (_Float16)u0.y;
        h[2] = (_Float16)u0.z; h[3] = (_Float16)u0.w;
        h[4] = (_Float16)u1.x; h[5] = (_Float16)u1.y;
        h[6] = (_Float16)u1.z; h[7] = (_Float16)u1.w;
        *(v8h*)&As[e] = h;
    }
    __syncthreads();

    const int ncol  = wave * 16 + (lane & 15);
    const int arow  = lane & 15;
    const int khalf = lane >> 4;   // 0: lanes 0-15, 1: lanes 16-31

    v8f c;
#pragma unroll
    for (int i = 0; i < 8; ++i) c[i] = BIAS ? bias[ncol] : 0.0f;

#pragma unroll
    for (int k0 = 0; k0 < FIN; k0 += 32) {
        // A fragment: contiguous 16-half run, k = 16*khalf + s (matches B).
        const _Float16* pa = &As[arow * FIN + k0 + (khalf ? 16 : 0)];
        v8h alo = *(const v8h*)(pa);
        v8h ahi = *(const v8h*)(pa + 8);
        v16h a;
#pragma unroll
        for (int i = 0; i < 8; ++i) { a[i] = alo[i]; a[i + 8] = ahi[i]; }

        // B fragment: contiguous 32B column slice of transposed weight.
        const _Float16* pb = Wt + ncol * FIN + k0 + (khalf ? 16 : 0);
        v16h b = *(const v16h*)pb;

        c = __builtin_amdgcn_wmma_f32_16x16x32_f16(false, a, false, b,
                                                   (short)0, c, false, false);
    }

    // --- Epilogue: C frag -> LDS, then linear coalesced b128 stores ---
    const int crow0 = khalf * 8;   // VGPR r -> row r + 8*khalf, col = ncol
#pragma unroll
    for (int r = 0; r < 8; ++r) Cs[(crow0 + r) * NT + ncol] = c[r];
    __syncthreads();

    float* Cb = C + row_base * NT;   // contiguous 16*NT-float region
#pragma unroll
    for (int c8 = 0; c8 < (16 * NT) / (nthreads * 8); ++c8) {
        int e = tid * ((16 * NT) / nthreads) + c8 * 8;
        float4 v0 = *(const float4*)&Cs[e];
        float4 v1 = *(const float4*)&Cs[e + 4];
        *(float4*)&Cb[e]     = v0;
        *(float4*)&Cb[e + 4] = v1;
    }
}

// ---------------------------------------------------------------------------
// Edge scatter: one wave per edge, 4 features/lane (128 total)
// agg[dst] += t[src] * dinv[src]*dinv[dst]
// ---------------------------------------------------------------------------
__global__ void scatter_kernel(const int* __restrict__ src, const int* __restrict__ dst,
                               const float* __restrict__ t, const float* __restrict__ dinv,
                               float* __restrict__ agg, int E) {
    long gid = (long)blockIdx.x * blockDim.x + threadIdx.x;
    int e    = (int)(gid >> 5);
    if (e >= E) return;
    int lane = (int)(gid & 31);
    int s = src[e], d = dst[e];
    float norm = dinv[s] * dinv[d];
    const float4 v = *(const float4*)(t + s * FIN + lane * 4);
    float* o = agg + d * FIN + lane * 4;
    atomicAdd(o + 0, v.x * norm);
    atomicAdd(o + 1, v.y * norm);
    atomicAdd(o + 2, v.z * norm);
    atomicAdd(o + 3, v.w * norm);
}

// ---------------------------------------------------------------------------
// Post-aggregation: h = relu(agg + t*dinv^2 (self-loop) + bias)   [in place ok]
// ---------------------------------------------------------------------------
__global__ void post_kernel(const float* __restrict__ agg, const float* __restrict__ t,
                            const float* __restrict__ dinv, const float* __restrict__ bias,
                            float* __restrict__ h, int total) {
    int idx = blockIdx.x * blockDim.x + threadIdx.x;
    if (idx >= total) return;
    int node = idx >> 7;
    int f    = idx & 127;
    float di = dinv[node];
    float v  = agg[idx] + t[idx] * di * di + bias[f];
    h[idx] = fmaxf(v, 0.0f);
}

// ---------------------------------------------------------------------------
static inline size_t alignup(size_t v) { return (v + 255) & ~(size_t)255; }

extern "C" void kernel_launch(void* const* d_in, const int* in_sizes, int n_in,
                              void* d_out, int out_size, void* d_ws, size_t ws_size,
                              hipStream_t stream) {
    const float* x  = (const float*)d_in[0];
    const int*   ei = (const int*)d_in[1];
    const float* W1 = (const float*)d_in[2];
    const float* b1 = (const float*)d_in[3];
    const float* W2 = (const float*)d_in[4];
    const float* b2 = (const float*)d_in[5];
    const float* Wf = (const float*)d_in[6];
    const float* bf = (const float*)d_in[7];

    const int Nn = in_sizes[0] / FIN;   // 100000 (multiple of 16)
    const int E  = in_sizes[1] / 2;     // 1600000
    const int OUTC = 64;
    const int* src = ei;
    const int* dst = ei + E;

    // Workspace carve-up
    char* p = (char*)d_ws;
    float* bufA = (float*)p;            p += alignup((size_t)Nn * FIN * 4);
    float* bufB = (float*)p;            p += alignup((size_t)Nn * FIN * 4);
    float* dinv = (float*)p;            p += alignup((size_t)Nn * 4);
    _Float16* Wt1 = (_Float16*)p;       p += alignup((size_t)FIN * FIN * 2);
    _Float16* Wt2 = (_Float16*)p;       p += alignup((size_t)FIN * FIN * 2);
    _Float16* Wtf = (_Float16*)p;       p += alignup((size_t)FIN * OUTC * 2);

    // ---- Weight prep (f16, transposed) ----
    wconv_kernel<<<(FIN * FIN + 255) / 256, 256, 0, stream>>>(W1, Wt1, FIN, FIN);
    wconv_kernel<<<(FIN * FIN + 255) / 256, 256, 0, stream>>>(W2, Wt2, FIN, FIN);
    wconv_kernel<<<(FIN * OUTC + 255) / 256, 256, 0, stream>>>(Wf, Wtf, FIN, OUTC);

    // ---- Degrees -> dinv ----
    deg_init_kernel<<<(Nn + 255) / 256, 256, 0, stream>>>(dinv, Nn);
    deg_count_kernel<<<(E + 255) / 256, 256, 0, stream>>>(dst, dinv, E);
    deg_rsqrt_kernel<<<(Nn + 255) / 256, 256, 0, stream>>>(dinv, Nn);

    const int mblocks  = Nn / 16;                          // 6250 (exact)
    const int total    = Nn * FIN;                         // 12.8M, fits int
    const int featblks = (total + 255) / 256;
    const int edgeblks = (int)(((long)E * 32 + 255) / 256);

    // ---- Layer 1: t = x@W1 ; agg ; h1 = relu(agg + self + b1) ----
    gemm_wmma_kernel<FIN, false><<<mblocks, 256, 0, stream>>>(x, Wt1, nullptr, bufA);
    hipMemsetAsync(bufB, 0, (size_t)Nn * FIN * 4, stream);
    scatter_kernel<<<edgeblks, 256, 0, stream>>>(src, dst, bufA, dinv, bufB, E);
    post_kernel<<<featblks, 256, 0, stream>>>(bufB, bufA, dinv, b1, bufB, total);

    // ---- Layer 2: t2 = h1@W2 ; agg ; h2 = relu(agg + self + b2) ----
    gemm_wmma_kernel<FIN, false><<<mblocks, 256, 0, stream>>>(bufB, Wt2, nullptr, bufA);
    hipMemsetAsync(bufB, 0, (size_t)Nn * FIN * 4, stream);
    scatter_kernel<<<edgeblks, 256, 0, stream>>>(src, dst, bufA, dinv, bufB, E);
    post_kernel<<<featblks, 256, 0, stream>>>(bufB, bufA, dinv, b2, bufB, total);

    // ---- Final: out = h2 @ Wf + bf ----
    gemm_wmma_kernel<OUTC, true><<<mblocks, 128, 0, stream>>>(bufB, Wtf, bf,
                                                              (float*)d_out);
}
